// GraphModel_88364657148326
// MI455X (gfx1250) — compile-verified, matrix operand order checked by reading, not run on hardware
//
#include <hip/hip_runtime.h>

#define NFEAT   140
#define EMB     128
#define K0REAL  268
#define XSTRIDE 288   // K0 padded to 9 k-tiles of 32
#define GF      128
#define CONF    24

typedef __attribute__((ext_vector_type(16))) __bf16 v16bf;
typedef __attribute__((ext_vector_type(8)))  float  v8f;

__device__ __forceinline__ unsigned short f2bf(float f) {
  unsigned int u = __float_as_uint(f);
  u += 0x7FFFu + ((u >> 16) & 1u);          // round-to-nearest-even
  return (unsigned short)(u >> 16);
}

// Guaranteed-hardware f32 atomic add (no-return form, STOREcnt-tracked).
// Avoids any chance of a CAS-loop lowering for the 128M-update edge scatter.
__device__ __forceinline__ void atomAddF32(float* p, float v) {
  asm volatile("global_atomic_add_f32 %0, %1, off" : : "v"(p), "v"(v) : "memory");
}

// ---------------- degree / normalization ----------------
__global__ void k_fill(float* p, float v, int n) {
  int i = blockIdx.x * 256 + threadIdx.x;
  if (i < n) p[i] = v;
}
__global__ void k_deg(const int* __restrict__ ei, float* deg, int E) {
  int e = blockIdx.x * 256 + threadIdx.x;
  if (e < E) atomAddF32(&deg[ei[E + e]], 1.0f);
}
__global__ void k_rsqrt(float* p, int n) {
  int i = blockIdx.x * 256 + threadIdx.x;
  if (i < n) p[i] = rsqrtf(p[i]);
}

// ---------------- build X0 = [node_features | emb[opcode] | 0 pad] in bf16 ----------------
__global__ void k_build_x(const float* __restrict__ nf, const int* __restrict__ opc,
                          const float* __restrict__ emb, unsigned short* __restrict__ X, int n) {
  int t = blockIdx.x * 256 + threadIdx.x;
  if (t >= n * XSTRIDE) return;
  int i = t / XSTRIDE, f = t - i * XSTRIDE;
  float v = 0.0f;
  if (f < NFEAT)       v = nf[i * NFEAT + f];
  else if (f < K0REAL) v = emb[opc[i] * EMB + (f - NFEAT)];
  X[t] = f2bf(v);
}

// ---------------- pack W [Kreal, nOut] f32 -> per-fragment bf16 ----------------
// element t = ((kt*nTiles + nt)*512 + lane*16 + e); per-lane 32B contiguous.
__global__ void k_pack_w(const float* __restrict__ W, unsigned short* __restrict__ Wpk,
                         int Kreal, int nOut) {
  int nTiles = nOut >> 4;
  int frag = blockIdx.x;              // kt*nTiles + nt
  int li   = threadIdx.x;             // 0..511
  int lane = li >> 4, e = li & 15;
  int kt = frag / nTiles, nt = frag - kt * nTiles;
  int K = kt * 32 + (e >> 3) * 16 + (lane >> 4) * 8 + (e & 7);
  int N = nt * 16 + (lane & 15);
  float v = (K < Kreal) ? W[K * nOut + N] : 0.0f;
  Wpk[frag * 512 + li] = f2bf(v);
}

// ---------------- WMMA GEMM: H[N,nOut] = X[N,K] @ W[K,nOut] ----------------
// wave32; each wave computes a 16(M) x 64(N) strip with 4 accumulator fragments.
__global__ void k_gemm(const unsigned short* __restrict__ X,
                       const unsigned short* __restrict__ Wpk,
                       float* __restrict__ H,
                       int mTiles, int kTiles, int nOut) {
  int lane = threadIdx.x & 31;
  int wave = threadIdx.x >> 5;
  int nTiles  = nOut >> 4;
  int nGroups = nTiles >> 2;          // 64-col groups: 4 (nOut=256) or 2 (nOut=128)
  int mPerBlk = 8 / nGroups;
  int mi = wave / nGroups;
  int ng = wave - mi * nGroups;
  int mtile = blockIdx.x * mPerBlk + mi;
  if (mtile >= mTiles) return;        // wave-uniform: EXEC stays all-ones for WMMA

  int hl  = lane >> 4;                // half-wave id
  int row = mtile * 16 + (lane & 15);
  const unsigned short* xrow = X + row * XSTRIDE;

  v8f acc0 = {}, acc1 = {}, acc2 = {}, acc3 = {};
  for (int kt = 0; kt < kTiles; ++kt) {
    union { uint4 u[2]; v16bf v; } a;
    int kb = kt * 32 + hl * 8;
    a.u[0] = *(const uint4*)(xrow + kb);        // K = kb .. kb+7
    a.u[1] = *(const uint4*)(xrow + kb + 16);   // K = kb+16 .. kb+23
    const unsigned short* wb = Wpk + ((kt * nTiles + ng * 4) * 512) + lane * 16;
    union { uint4 u[2]; v16bf v; } b;
    b.u[0] = *(const uint4*)(wb);        b.u[1] = *(const uint4*)(wb + 8);
    acc0 = __builtin_amdgcn_wmma_f32_16x16x32_bf16(false, a.v, false, b.v, (short)0, acc0, false, false);
    b.u[0] = *(const uint4*)(wb + 512);  b.u[1] = *(const uint4*)(wb + 520);
    acc1 = __builtin_amdgcn_wmma_f32_16x16x32_bf16(false, a.v, false, b.v, (short)0, acc1, false, false);
    b.u[0] = *(const uint4*)(wb + 1024); b.u[1] = *(const uint4*)(wb + 1032);
    acc2 = __builtin_amdgcn_wmma_f32_16x16x32_bf16(false, a.v, false, b.v, (short)0, acc2, false, false);
    b.u[0] = *(const uint4*)(wb + 1536); b.u[1] = *(const uint4*)(wb + 1544);
    acc3 = __builtin_amdgcn_wmma_f32_16x16x32_bf16(false, a.v, false, b.v, (short)0, acc3, false, false);
  }
  // C/D layout: VGPR r -> M = r + 8*hl ; lane%16 -> N
  int colBase = ng * 64 + (lane & 15);
  int rowBase = mtile * 16 + hl * 8;
#pragma unroll
  for (int r = 0; r < 8; ++r) {
    float* o = H + (rowBase + r) * nOut + colBase;
    o[0]  = acc0[r];
    o[16] = acc1[r];
    o[32] = acc2[r];
    o[48] = acc3[r];
  }
}

// ---------------- aggregation ----------------
__global__ void k_agg_init(const float* __restrict__ H, const float* __restrict__ dis,
                           const float* __restrict__ bias, float* __restrict__ OUT,
                           int n, int nOut) {
  int t = blockIdx.x * 256 + threadIdx.x;
  if (t >= n * nOut) return;
  int i = t / nOut, f = t - i * nOut;
  float w = dis[i];
  OUT[t] = H[t] * w * w + bias[f];
}
__global__ void k_agg_edge(const int* __restrict__ ei, const float* __restrict__ H,
                           const float* __restrict__ dis, float* __restrict__ OUT,
                           int E, int nOut) {
  int e = blockIdx.x;
  int f = threadIdx.x;
  int s = ei[e], d = ei[E + e];
  float w = dis[s] * dis[d];
  atomAddF32(&OUT[d * nOut + f], H[s * nOut + f] * w);
}
__global__ void k_relu_bf(const float* __restrict__ OUT, unsigned short* __restrict__ X,
                          int n, int nOut) {
  int t = blockIdx.x * 256 + threadIdx.x;
  if (t >= n * nOut) return;
  int i = t / nOut, f = t - i * nOut;
  float v = OUT[t];
  X[i * XSTRIDE + f] = f2bf(v > 0.0f ? v : 0.0f);
}

// ---------------- graph mean + MLP head ----------------
__global__ void k_zero(float* p, int n) {
  int i = blockIdx.x * 128 + threadIdx.x;
  if (i < n) p[i] = 0.0f;
}
__global__ void k_mean(const float* __restrict__ OUT, float* __restrict__ xsum,
                       int rowsPerBlock, int n) {
  int f = threadIdx.x;                 // 128
  int r0 = blockIdx.x * rowsPerBlock;
  int r1 = r0 + rowsPerBlock; if (r1 > n) r1 = n;
  float s = 0.0f;
  for (int r = r0; r < r1; ++r) {
    float v = OUT[r * GF + f];
    s += (v > 0.0f ? v : 0.0f);
  }
  atomAddF32(&xsum[f], s);
}
__global__ void k_head1(const float* __restrict__ cfg, const float* __restrict__ xsum,
                        const float* __restrict__ W, const float* __restrict__ bias,
                        float* __restrict__ H1, float invN) {
  int c = blockIdx.x, f = threadIdx.x; // 128
  float s = bias[f];
  for (int k = 0; k < CONF; ++k) s += cfg[c * CONF + k] * W[k * GF + f];
  for (int k = 0; k < GF; ++k)   s += xsum[k] * invN * W[(CONF + k) * GF + f];
  H1[c * GF + f] = s > 0.0f ? s : 0.0f;
}
__global__ void k_head2(const float* __restrict__ H1, const float* __restrict__ W,
                        const float* __restrict__ bias, float* __restrict__ H2) {
  int c = blockIdx.x, f = threadIdx.x;
  float s = bias[f];
  for (int k = 0; k < GF; ++k) s += H1[c * GF + k] * W[k * GF + f];
  H2[c * GF + f] = s > 0.0f ? s : 0.0f;
}
__global__ void k_head3(const float* __restrict__ H2, const float* __restrict__ W,
                        const float* __restrict__ bias, float* __restrict__ out, int C) {
  int c = blockIdx.x * 256 + threadIdx.x;
  if (c >= C) return;
  float s = bias[0];
  for (int k = 0; k < GF; ++k) s += H2[c * GF + k] * W[k];
  out[c] = s;
}

extern "C" void kernel_launch(void* const* d_in, const int* in_sizes, int n_in,
                              void* d_out, int out_size, void* d_ws, size_t ws_size,
                              hipStream_t stream) {
  const float* config = (const float*)d_in[0];
  const float* nf     = (const float*)d_in[1];
  const int*   opc    = (const int*)d_in[2];
  const int*   ei     = (const int*)d_in[3];
  const float* emb    = (const float*)d_in[4];
  const float* W0 = (const float*)d_in[5];  const float* b0 = (const float*)d_in[6];
  const float* W1 = (const float*)d_in[7];  const float* b1 = (const float*)d_in[8];
  const float* W2 = (const float*)d_in[9];  const float* b2 = (const float*)d_in[10];
  const float* D1w = (const float*)d_in[11]; const float* D1b = (const float*)d_in[12];
  const float* D2w = (const float*)d_in[13]; const float* D2b = (const float*)d_in[14];
  const float* D3w = (const float*)d_in[15]; const float* D3b = (const float*)d_in[16];
  float* out = (float*)d_out;

  const int N = in_sizes[2];        // 50000 nodes
  const int E = in_sizes[3] / 2;    // 500000 edges
  const int C = in_sizes[0] / CONF; // 4096 configs

  // workspace layout
  char* ws = (char*)d_ws;
  size_t off = 0;
  auto alloc = [&](size_t bytes) -> void* {
    void* p = ws + off;
    off = (off + bytes + 255) & ~(size_t)255;
    return p;
  };
  float*          dis  = (float*)alloc((size_t)N * 4);
  unsigned short* X    = (unsigned short*)alloc((size_t)N * XSTRIDE * 2);
  float*          Hbuf = (float*)alloc((size_t)N * 256 * 4);
  float*          OUT  = (float*)alloc((size_t)N * 256 * 4);
  unsigned short* W0p  = (unsigned short*)alloc((size_t)288 * 256 * 2);
  unsigned short* W1p  = (unsigned short*)alloc((size_t)256 * 256 * 2);
  unsigned short* W2p  = (unsigned short*)alloc((size_t)256 * 128 * 2);
  float*          xsum = (float*)alloc((size_t)GF * 4);
  float*          Hc1  = (float*)alloc((size_t)C * GF * 4);
  float*          Hc2  = (float*)alloc((size_t)C * GF * 4);
  (void)ws_size; (void)n_in; (void)out_size;

  // degrees (self loop baked in as 1.0) -> deg^-1/2 in place
  k_fill <<<(N + 255) / 256, 256, 0, stream>>>(dis, 1.0f, N);
  k_deg  <<<(E + 255) / 256, 256, 0, stream>>>(ei, dis, E);
  k_rsqrt<<<(N + 255) / 256, 256, 0, stream>>>(dis, N);

  // input features + weight packing
  k_build_x<<<(N * XSTRIDE + 255) / 256, 256, 0, stream>>>(nf, opc, emb, X, N);
  k_pack_w<<<(288 / 32) * (256 / 16), 512, 0, stream>>>(W0, W0p, 268, 256);
  k_pack_w<<<(256 / 32) * (256 / 16), 512, 0, stream>>>(W1, W1p, 256, 256);
  k_pack_w<<<(256 / 32) * (128 / 16), 512, 0, stream>>>(W2, W2p, 256, 128);

  const int mT = (N + 15) / 16;     // 3125

  // ---- layer 0: K=288(pad of 268), nOut=256
  k_gemm    <<<(mT + 1) / 2, 256, 0, stream>>>(X, W0p, Hbuf, mT, 9, 256);
  k_agg_init<<<(N * 256 + 255) / 256, 256, 0, stream>>>(Hbuf, dis, b0, OUT, N, 256);
  k_agg_edge<<<E, 256, 0, stream>>>(ei, Hbuf, dis, OUT, E, 256);
  k_relu_bf <<<(N * 256 + 255) / 256, 256, 0, stream>>>(OUT, X, N, 256);

  // ---- layer 1: K=256, nOut=256
  k_gemm    <<<(mT + 1) / 2, 256, 0, stream>>>(X, W1p, Hbuf, mT, 8, 256);
  k_agg_init<<<(N * 256 + 255) / 256, 256, 0, stream>>>(Hbuf, dis, b1, OUT, N, 256);
  k_agg_edge<<<E, 256, 0, stream>>>(ei, Hbuf, dis, OUT, E, 256);
  k_relu_bf <<<(N * 256 + 255) / 256, 256, 0, stream>>>(OUT, X, N, 256);

  // ---- layer 2: K=256, nOut=128 (relu fused into mean reduce)
  k_gemm    <<<(mT + 3) / 4, 256, 0, stream>>>(X, W2p, Hbuf, mT, 8, 128);
  k_agg_init<<<(N * 128 + 255) / 256, 256, 0, stream>>>(Hbuf, dis, b2, OUT, N, 128);
  k_agg_edge<<<E, 128, 0, stream>>>(ei, Hbuf, dis, OUT, E, 128);

  // graph mean + head
  k_zero <<<1, 128, 0, stream>>>(xsum, GF);
  k_mean <<<(N + 199) / 200, 128, 0, stream>>>(OUT, xsum, 200, N);
  k_head1<<<C, 128, 0, stream>>>(config, xsum, D1w, D1b, Hc1, 1.0f / (float)N);
  k_head2<<<C, 128, 0, stream>>>(Hc1, D2w, D2b, Hc2);
  k_head3<<<(C + 255) / 256, 256, 0, stream>>>(Hc2, D3w, D3b, out, C);
}
